// LANScoreNet_80444737454190
// MI455X (gfx1250) — matrix operand order repaired and checked by reading, not compiled, not run on hardware
//
#include <hip/hip_runtime.h>
#include <hip/hip_bf16.h>

// ---------------------------------------------------------------------------
// Problem constants (from setup_inputs): B=4, N=8192, F=H=128, L=4, K=4, 30 steps
// ---------------------------------------------------------------------------
#define BB     4
#define NN     8192
#define HH     128
#define FF     128
#define LL     4
#define KK     4
#define NSTEPS 30

typedef __attribute__((ext_vector_type(16))) __bf16 v16bf;
typedef __attribute__((ext_vector_type(8)))  float  v8f;
typedef __attribute__((ext_vector_type(4)))  unsigned int u32x4;
typedef __attribute__((ext_vector_type(8)))  int  i32x8;
typedef __attribute__((ext_vector_type(4)))  int  i32x4;

// d_ws layout (bytes)
#define PCL_BYTES   (BB * NN * 3 * 4)          // 393216
#define OFF_PCL     0
#define OFF_ACC     (PCL_BYTES)                // 393216
#define OFF_IDX     (2 * PCL_BYTES)            // 786432  (B*N*K int = 524288 B)
#define OFF_WPACK   (OFF_IDX + BB * NN * KK * 4)         // 1310720 (9*128*128 bf16)
#define OFF_LAT     (OFF_WPACK + 9 * HH * HH * 2)        // 1605632 (B*N*H f32)

#define SLOT_SZ     (HH * HH)                  // 16384 bf16 per matrix

static __device__ inline void lds_fence() {
  // CDNA5 split counter: wait for all outstanding LDS ops from this wave.
  asm volatile("s_wait_dscnt 0" ::: "memory");
}

// ---------------------------------------------------------------------------
// Tensor Data Mover: 1-D bulk copy global -> LDS.
// D# per CDNA5 ISA ch.8: group0 {count=1, lds_addr, global_addr, type=2},
// group1 {data_size=8B, tensor_dim0=tile_dim0=n_qwords, stride=n_qwords}.
// Issued by one wave; completion tracked on that wave's TENSORcnt.
// ---------------------------------------------------------------------------
static __device__ inline void tdm_load_1d(unsigned lds_addr, const void* gptr,
                                          unsigned n_qwords) {
  const unsigned long long ga = (unsigned long long)gptr;
  u32x4 g0;
  g0[0] = 1u;                                             // count=1, user mode
  g0[1] = lds_addr;                                       // lds_addr [63:32]
  g0[2] = (unsigned)(ga & 0xFFFFFFFFu);                   // global_addr lo
  g0[3] = (unsigned)((ga >> 32) & 0x01FFFFFFu) | (2u << 30); // addr hi | type=2
  i32x8 g1;
  g1[0] = (int)(3u << 16);                                // data_size=3 (8B)
  g1[1] = (int)((n_qwords & 0xFFFFu) << 16);              // tensor_dim0 lo16
  g1[2] = (int)(((n_qwords >> 16) & 0xFFFFu) | (1u << 16)); // dim0 hi | dim1=1
  g1[3] = (int)((n_qwords & 0xFFFFu) << 16);              // tile_dim0
  g1[4] = 1;                                              // tile_dim1=1
  g1[5] = (int)n_qwords;                                  // dim0_stride lo32
  g1[6] = 0;
  g1[7] = 0;
  i32x4 z = {0, 0, 0, 0};
#if defined(__clang_major__) && (__clang_major__ >= 23)
  i32x8 z8 = {0, 0, 0, 0, 0, 0, 0, 0};
  __builtin_amdgcn_tensor_load_to_lds(g0, g1, z, z, z8, 0);
#else
  __builtin_amdgcn_tensor_load_to_lds(g0, g1, z, z, 0);
#endif
}

static __device__ inline v8f wmma_bf16(v16bf a, v16bf b, v8f c) {
  return __builtin_amdgcn_wmma_f32_16x16x32_bf16(
      /*neg_a=*/false, a, /*neg_b=*/false, b,
      /*c_mod=*/(short)0, c, /*reuse_a=*/false, /*reuse_b=*/false);
}

// A-operand (16x32 bf16, wave32 layout) built from a row-major f32 row.
// lane: m = lane&15, hi = lane>>4.  elems 0..7 -> K = tk*32 + 8*hi + e,
// elems 8..15 -> K = tk*32 + 16 + 8*hi + e.
static __device__ inline v16bf load_A_f32(const float* rowp, int tk, int hi) {
  v16bf a;
  const int k0 = tk * 32 + 8 * hi;
#pragma unroll
  for (int e = 0; e < 8; ++e) {
    a[e]     = (__bf16)rowp[k0 + e];
    a[8 + e] = (__bf16)rowp[k0 + 16 + e];
  }
  return a;
}

// Same A-layout but from a bf16 LDS stage [16][128].
static __device__ inline v16bf load_A_lds(const __bf16* st, int tk, int lm, int hi) {
  const __bf16* p0 = st + lm * HH + tk * 32 + 8 * hi;
  v16bf a;
#pragma unroll
  for (int e = 0; e < 8; ++e) { a[e] = p0[e]; a[8 + e] = p0[16 + e]; }
  return a;
}

// B-operand (32x16 bf16): lane holds N column = nt*16 + (lane&15);
// elems 0..7 -> K = tk*32 + 8*hi + e, elems 8..15 -> K = tk*32 + 16 + 8*hi + e.
static __device__ inline v16bf load_B(const __bf16* wslot, int tk, int nt, int lm, int hi) {
  const __bf16* p0 = wslot + (tk * 32 + 8 * hi) * HH + nt * 16 + lm;
  v16bf b;
#pragma unroll
  for (int e = 0; e < 8; ++e) { b[e] = p0[e * HH]; b[8 + e] = p0[(16 + e) * HH]; }
  return b;
}

// ---------------------------------------------------------------------------
// prep: repack weights to bf16 (B-operand friendly, i.e. transposed [k][g]),
//       copy pcl_noisy -> pcl_cur, zero acc.
// ---------------------------------------------------------------------------
#define N_WELEM (9 * SLOT_SZ)                  // 147456
#define N_PREP  (N_WELEM + 2 * BB * NN * 3)    // + copy + zero

__global__ void prep_kernel(const float* __restrict__ pcl_noisy,
                            const float* __restrict__ fc0_w,
                            const float* __restrict__ fc1_w,
                            const float* __restrict__ conv_p_w,
                            float* __restrict__ pcl_cur,
                            float* __restrict__ acc,
                            __bf16* __restrict__ wpack) {
  for (int t = blockIdx.x * blockDim.x + threadIdx.x; t < N_PREP;
       t += gridDim.x * blockDim.x) {
    if (t < N_WELEM) {
      const int s = t >> 14;          // matrix slot
      const int e = t & (SLOT_SZ - 1);
      const int k = e >> 7;           // input channel
      const int g = e & 127;          // output channel
      float v;
      if (s < 4)      v = fc0_w[(s * HH + g) * HH + k];
      else if (s < 8) v = fc1_w[((s - 4) * HH + g) * HH + k];
      else            v = conv_p_w[g * (3 + FF) + 3 + k];   // lat part of conv_p
      wpack[t] = (__bf16)v;
    } else {
      const int u = t - N_WELEM;
      if (u < BB * NN * 3) pcl_cur[u] = pcl_noisy[u];
      else                 acc[u - BB * NN * 3] = 0.0f;
    }
  }
}

// ---------------------------------------------------------------------------
// lat = feat @ Wp_lat^T + conv_p_b   (step-invariant; WMMA bf16, f32 accum)
// One wave per 16-row tile; M = B*N = 32768 -> 2048 waves.
// Slot-8 weight matrix staged to LDS via TDM, shared by the 4 waves.
// ---------------------------------------------------------------------------
__global__ void lat_kernel(const float* __restrict__ feat,
                           const __bf16* __restrict__ wpack,
                           const float* __restrict__ conv_p_b,
                           float* __restrict__ lat) {
  __shared__ __bf16 wl_lds[SLOT_SZ];            // 32 KB
  const int tid  = threadIdx.x;
  if ((tid >> 5) == 0) {                        // wave 0 issues the DMA
    tdm_load_1d((unsigned)(uintptr_t)wl_lds, wpack + 8 * SLOT_SZ,
                SLOT_SZ * 2 / 8);
    __builtin_amdgcn_s_wait_tensorcnt(0);
  }
  __syncthreads();

  const int gt   = blockIdx.x * blockDim.x + tid;
  const int wv   = gt >> 5;
  const int lane = tid & 31;
  const int lm   = lane & 15, hi = lane >> 4;
  const int row0 = wv * 16;

  const float* arow = feat + (size_t)(row0 + lm) * FF;
  v16bf A[4];
#pragma unroll
  for (int tk = 0; tk < 4; ++tk) A[tk] = load_A_f32(arow, tk, hi);

#pragma unroll
  for (int nt = 0; nt < 8; ++nt) {
    const int col = lm + nt * 16;
    v8f c;
    const float bb = conv_p_b[col];
#pragma unroll
    for (int j = 0; j < 8; ++j) c[j] = bb;
#pragma unroll
    for (int tk = 0; tk < 4; ++tk) c = wmma_bf16(A[tk], load_B(wl_lds, tk, nt, lm, hi), c);
    // C layout: vgpr j -> row m = j + 8*hi, column = col
#pragma unroll
    for (int j = 0; j < 8; ++j)
      lat[(size_t)(row0 + j + 8 * hi) * HH + col] = c[j];
  }
}

// ---------------------------------------------------------------------------
// KNN: top-4 nearest current-cloud points per fixed query (per batch).
// Block = 256 threads (one query each), cloud tiled through LDS.
// ---------------------------------------------------------------------------
__global__ void knn_kernel(const float* __restrict__ pcl_noisy,
                           const float* __restrict__ pcl_cur,
                           int* __restrict__ idx) {
  __shared__ float pts[256 * 4];   // padded to 4 to dodge bank conflicts
  const int gq = blockIdx.x * 256 + threadIdx.x;   // 0 .. B*N-1
  const int b  = gq >> 13;
  const float* q = pcl_noisy + (size_t)gq * 3;
  const float qx = q[0], qy = q[1], qz = q[2];
  const float* cloud = pcl_cur + (size_t)b * NN * 3;

  float bd0 = 3e38f, bd1 = 3e38f, bd2 = 3e38f, bd3 = 3e38f;
  int   bi0 = 0, bi1 = 0, bi2 = 0, bi3 = 0;

  for (int base = 0; base < NN; base += 256) {
    __syncthreads();
    const int t = threadIdx.x;
    pts[t * 4 + 0] = cloud[(base + t) * 3 + 0];
    pts[t * 4 + 1] = cloud[(base + t) * 3 + 1];
    pts[t * 4 + 2] = cloud[(base + t) * 3 + 2];
    __syncthreads();
    for (int j = 0; j < 256; ++j) {
      const float dx = pts[j * 4 + 0] - qx;
      const float dy = pts[j * 4 + 1] - qy;
      const float dz = pts[j * 4 + 2] - qz;
      const float d  = fmaf(dx, dx, fmaf(dy, dy, dz * dz));
      const int pid  = base + j;
      if (d < bd3) {
        if (d < bd0)      { bd3=bd2;bi3=bi2; bd2=bd1;bi2=bi1; bd1=bd0;bi1=bi0; bd0=d;bi0=pid; }
        else if (d < bd1) { bd3=bd2;bi3=bi2; bd2=bd1;bi2=bi1; bd1=d;  bi1=pid; }
        else if (d < bd2) { bd3=bd2;bi3=bi2; bd2=d;  bi2=pid; }
        else              { bd3=d;  bi3=pid; }
      }
    }
  }
  int* o = idx + (size_t)gq * KK;
  o[0] = bi0; o[1] = bi1; o[2] = bi2; o[3] = bi3;
}

// ---------------------------------------------------------------------------
// Fused score network: gather + seed + 4 ResNet blocks (8 WMMA GEMMs) + head
// + atomic scatter.  Block = 128 threads = 4 waves; all 8 fc weight matrices
// (256 KB bf16) are pulled into LDS by one TDM descriptor; per-wave 8 KB stage.
// grid = 256 blocks -> 1024 waves x 8 tiles = 8192 tiles of 16 rows.
// ---------------------------------------------------------------------------
#define SM_W      (8 * SLOT_SZ * 2)            // 262144 B weights
#define SM_STAGE  (16 * HH * 4)                // 8192 B per-wave stage
#define SM_META   384                          // per-wave meta
#define SM_TOTAL  (SM_W + 4 * SM_STAGE + 4 * SM_META)   // 296448 B (< 320 KB WGP LDS)

__global__ void score_kernel(const float* __restrict__ pcl_noisy,
                             const float* __restrict__ pcl_cur,
                             const float* __restrict__ lat,
                             const int*   __restrict__ idx,
                             const __bf16* __restrict__ wpack,
                             float* __restrict__ acc,
                             const float* __restrict__ conv_p_w,
                             const float* __restrict__ bn0_scale,
                             const float* __restrict__ bn0_shift,
                             const float* __restrict__ bn1_scale,
                             const float* __restrict__ bn1_shift,
                             const float* __restrict__ fc0_b,
                             const float* __restrict__ fc1_b,
                             const float* __restrict__ bn_out_scale,
                             const float* __restrict__ bn_out_shift,
                             const float* __restrict__ conv_out_w,
                             const float* __restrict__ conv_out_b) {
  extern __shared__ char smem[];
  __bf16* wlds = (__bf16*)smem;

  const int tid = threadIdx.x;
  const int wv   = tid >> 5;
  const int lane = tid & 31;
  const int lm   = lane & 15, hi = lane >> 4;

  // One TDM descriptor moves all 8 fc matrices (256 KB) into LDS.
  if (wv == 0) {
    tdm_load_1d((unsigned)(uintptr_t)wlds, wpack, SM_W / 8);
    __builtin_amdgcn_s_wait_tensorcnt(0);
  }
  __syncthreads();

  float*  stageF = (float*)(smem + SM_W + wv * SM_STAGE);
  __bf16* stageH = (__bf16*)stageF;
  char*   metaB  = smem + SM_W + 4 * SM_STAGE + wv * SM_META;
  int*    mBN  = (int*)metaB;            // 16 x (b*N + n)
  int*    mOut = (int*)(metaB + 64);     // 16 x (b*N + neighbor id)
  float*  mX   = (float*)(metaB + 128);  // 16 x 3 centered coords

  const int waveGlobal = blockIdx.x * 4 + wv;   // 0..1023

#pragma unroll 1
  for (int t = 0; t < 8; ++t) {
    const int tile = t * 1024 + waveGlobal;     // 0..8191
    const int r0   = tile * 16;
    const int b    = r0 >> 15;                  // / (N*K)

    if (lane < 16) {                            // one meta row per lane
      const int r  = r0 + lane;
      const int n  = (r >> 2) & (NN - 1);
      const int k  = r & (KK - 1);
      const int bn = (b << 13) + n;
      const int id = idx[(size_t)bn * KK + k];
      const int bp = (b << 13) + id;
      mBN[lane]  = bn;
      mOut[lane] = bp;
      mX[lane * 3 + 0] = pcl_cur[bp * 3 + 0] - pcl_noisy[bn * 3 + 0];
      mX[lane * 3 + 1] = pcl_cur[bp * 3 + 1] - pcl_noisy[bn * 3 + 1];
      mX[lane * 3 + 2] = pcl_cur[bp * 3 + 2] - pcl_noisy[bn * 3 + 2];
    }
    lds_fence();

    // hoist per-row meta into registers (rows m = j + 8*hi)
    float xr[8][3];
    int   bnr[8];
#pragma unroll
    for (int j = 0; j < 8; ++j) {
      const int m = j + 8 * hi;
      xr[j][0] = mX[m * 3 + 0]; xr[j][1] = mX[m * 3 + 1]; xr[j][2] = mX[m * 3 + 2];
      bnr[j]   = mBN[m];
    }

    // h0 = x . Wxyz(col) + lat[bn][col]   (C-layout registers)
    v8f h[8];
#pragma unroll
    for (int nt = 0; nt < 8; ++nt) {
      const int col = lm + nt * 16;
      const float w0 = conv_p_w[col * (3 + FF) + 0];
      const float w1 = conv_p_w[col * (3 + FF) + 1];
      const float w2 = conv_p_w[col * (3 + FF) + 2];
#pragma unroll
      for (int j = 0; j < 8; ++j) {
        const float lv = lat[(size_t)bnr[j] * HH + col];
        h[nt][j] = fmaf(xr[j][0], w0, fmaf(xr[j][1], w1, fmaf(xr[j][2], w2, lv)));
      }
    }

    // 4 ResNet blocks, two bf16 WMMA GEMMs each, residual in f32 accumulators
#pragma unroll
    for (int i = 0; i < LL; ++i) {
      // a = relu(h*s0+b0) -> bf16 stage
#pragma unroll
      for (int nt = 0; nt < 8; ++nt) {
        const int col = lm + nt * 16;
        const float s0 = bn0_scale[i * HH + col];
        const float c0 = bn0_shift[i * HH + col];
#pragma unroll
        for (int j = 0; j < 8; ++j) {
          float a = fmaf(h[nt][j], s0, c0);
          a = a > 0.0f ? a : 0.0f;
          stageH[(j + 8 * hi) * HH + col] = (__bf16)a;
        }
      }
      lds_fence();
      v16bf A0[4];
#pragma unroll
      for (int tk = 0; tk < 4; ++tk) A0[tk] = load_A_lds(stageH, tk, lm, hi);
      lds_fence();

      // net = a @ fc0^T + b0;  a2 = relu(net*s1+b1) -> stage (overwrite)
      const __bf16* w0s = wlds + i * SLOT_SZ;
#pragma unroll
      for (int nt = 0; nt < 8; ++nt) {
        const int col = lm + nt * 16;
        v8f c;
        const float bb = fc0_b[i * HH + col];
#pragma unroll
        for (int j = 0; j < 8; ++j) c[j] = bb;
#pragma unroll
        for (int tk = 0; tk < 4; ++tk) c = wmma_bf16(A0[tk], load_B(w0s, tk, nt, lm, hi), c);
        const float s1 = bn1_scale[i * HH + col];
        const float c1 = bn1_shift[i * HH + col];
#pragma unroll
        for (int j = 0; j < 8; ++j) {
          float a2 = fmaf(c[j], s1, c1);
          a2 = a2 > 0.0f ? a2 : 0.0f;
          stageH[(j + 8 * hi) * HH + col] = (__bf16)a2;
        }
      }
      lds_fence();
      v16bf A1[4];
#pragma unroll
      for (int tk = 0; tk < 4; ++tk) A1[tk] = load_A_lds(stageH, tk, lm, hi);
      lds_fence();

      // h = h + a2 @ fc1^T + b1   (residual seeded into WMMA C)
      const __bf16* w1s = wlds + (4 + i) * SLOT_SZ;
#pragma unroll
      for (int nt = 0; nt < 8; ++nt) {
        const int col = lm + nt * 16;
        v8f c = h[nt];
        const float bb = fc1_b[i * HH + col];
#pragma unroll
        for (int j = 0; j < 8; ++j) c[j] += bb;
#pragma unroll
        for (int tk = 0; tk < 4; ++tk) c = wmma_bf16(A1[tk], load_B(w1s, tk, nt, lm, hi), c);
        h[nt] = c;
      }
    }

    // head: aout = relu(h*so+sho); out[d] = aout . conv_out_w[d] + b[d]
#pragma unroll
    for (int nt = 0; nt < 8; ++nt) {
      const int col = lm + nt * 16;
      const float so = bn_out_scale[col];
      const float sh = bn_out_shift[col];
#pragma unroll
      for (int j = 0; j < 8; ++j) {
        float a = fmaf(h[nt][j], so, sh);
        stageF[(j + 8 * hi) * HH + col] = a > 0.0f ? a : 0.0f;
      }
    }
    lds_fence();
    if (lane < 16) {
      const int row = lane;
      float s0 = 0.0f, s1 = 0.0f, s2 = 0.0f;
      for (int c = 0; c < HH; ++c) {
        const float av = stageF[row * HH + c];
        s0 = fmaf(av, conv_out_w[0 * HH + c], s0);
        s1 = fmaf(av, conv_out_w[1 * HH + c], s1);
        s2 = fmaf(av, conv_out_w[2 * HH + c], s2);
      }
      const int o = mOut[row] * 3;
      atomicAdd(&acc[o + 0], s0 + conv_out_b[0]);
      atomicAdd(&acc[o + 1], s1 + conv_out_b[1]);
      atomicAdd(&acc[o + 2], s2 + conv_out_b[2]);
    }
    lds_fence();
  }
}

// ---------------------------------------------------------------------------
// update: pcl += step_size * decay^step * acc;  acc = 0
// ---------------------------------------------------------------------------
__global__ void update_kernel(float* __restrict__ pcl_cur,
                              float* __restrict__ acc,
                              const float* __restrict__ step_size,
                              const float* __restrict__ step_decay,
                              int step) {
  const int t = blockIdx.x * blockDim.x + threadIdx.x;
  if (t < BB * NN * 3) {
    const float s = step_size[0] * __powf(step_decay[0], (float)step);
    pcl_cur[t] = fmaf(s, acc[t], pcl_cur[t]);
    acc[t] = 0.0f;
  }
}

// ---------------------------------------------------------------------------
extern "C" void kernel_launch(void* const* d_in, const int* in_sizes, int n_in,
                              void* d_out, int out_size, void* d_ws, size_t ws_size,
                              hipStream_t stream) {
  const float* pcl_noisy    = (const float*)d_in[0];
  const float* feat         = (const float*)d_in[1];
  const float* conv_p_w     = (const float*)d_in[2];
  const float* conv_p_b     = (const float*)d_in[3];
  const float* fc0_w        = (const float*)d_in[4];
  const float* fc0_b        = (const float*)d_in[5];
  const float* fc1_w        = (const float*)d_in[6];
  const float* fc1_b        = (const float*)d_in[7];
  const float* bn0_scale    = (const float*)d_in[8];
  const float* bn0_shift    = (const float*)d_in[9];
  const float* bn1_scale    = (const float*)d_in[10];
  const float* bn1_shift    = (const float*)d_in[11];
  const float* bn_out_scale = (const float*)d_in[12];
  const float* bn_out_shift = (const float*)d_in[13];
  const float* conv_out_w   = (const float*)d_in[14];
  const float* conv_out_b   = (const float*)d_in[15];
  const float* step_size    = (const float*)d_in[16];
  const float* step_decay   = (const float*)d_in[17];

  char* ws = (char*)d_ws;
  float*  pcl_cur = (float*)(ws + OFF_PCL);
  float*  acc     = (float*)(ws + OFF_ACC);
  int*    idx     = (int*)(ws + OFF_IDX);
  __bf16* wpack   = (__bf16*)(ws + OFF_WPACK);
  float*  lat     = (float*)(ws + OFF_LAT);

  prep_kernel<<<1344, 256, 0, stream>>>(pcl_noisy, fc0_w, fc1_w, conv_p_w,
                                        pcl_cur, acc, wpack);
  lat_kernel<<<512, 128, 0, stream>>>(feat, wpack, conv_p_b, lat);

  for (int step = 0; step < NSTEPS; ++step) {
    knn_kernel<<<BB * NN / 256, 256, 0, stream>>>(pcl_noisy, pcl_cur, idx);
    score_kernel<<<256, 128, SM_TOTAL, stream>>>(
        pcl_noisy, pcl_cur, lat, idx, wpack, acc, conv_p_w,
        bn0_scale, bn0_shift, bn1_scale, bn1_shift, fc0_b, fc1_b,
        bn_out_scale, bn_out_shift, conv_out_w, conv_out_b);
    update_kernel<<<(BB * NN * 3 + 255) / 256, 256, 0, stream>>>(
        pcl_cur, acc, step_size, step_decay, step);
  }
  hipMemcpyAsync(d_out, pcl_cur, PCL_BYTES, hipMemcpyDeviceToDevice, stream);
}
